// SAE_22711787062021
// MI455X (gfx1250) — matrix operand order repaired and checked by reading, not compiled, not run on hardware
//
#include <hip/hip_runtime.h>
#include <hip/hip_bf16.h>
#include <stdint.h>

// ---------------- problem constants ----------------
#define TOK   16384
#define IN    1024
#define HID   16384
#define KTOP  32

// encode tiling: block = 256 thr (8 waves) covers 32 tokens x 256 hidden.
// waves arranged 2(M) x 4(N); each wave owns 1 A-tile x 4 B-tiles (4 WMMAs / A-frag).
// K is staged in 32-wide chunks, double-buffered in LDS with async copies.
#define MT    32      // tokens per block
#define NTB   256     // hidden per block-iteration
#define TN    4       // N tiles (16 cols each) per wave
#define KC    32      // K chunk staged per inner step
#define NCH   (IN / KC)
#define ASTR  40      // A LDS row stride (bf16), 80B = 5*16
#define BSTR  40      // B LDS row stride (bf16), 80B = 5*16
#define LSTR  260     // logits LDS row stride (floats)

typedef __attribute__((ext_vector_type(16))) __bf16  v16bf;
typedef __attribute__((ext_vector_type(8)))  float   v8f;
typedef __attribute__((ext_vector_type(4)))  unsigned int u32x4;
typedef __attribute__((ext_vector_type(4)))  int     i32x4;
typedef __attribute__((ext_vector_type(4)))  float   f32x4;

union Frag { v16bf v; u32x4 q[2]; };

struct alignas(8) bf16x4s { __bf16 v[4]; };

// ---------------- async-copy plumbing (gfx1250 path, guarded) -------------
#if defined(__HIP_DEVICE_COMPILE__) && __has_builtin(__builtin_amdgcn_global_load_async_to_lds_b128)
#define HAVE_ASYNC_LDS 1
__device__ __forceinline__ void async_cp16(const void* gsrc, void* lds_dst) {
  // builtin signature (hipcc-verified): (v4i AS1*, v4i AS3*, imm, imm)
  // flat LDS address truncates to the AS3 offset (ISA 10.2: low 32 bits are used)
  auto g = (__attribute__((address_space(1))) i32x4*)(unsigned long long)gsrc;
  auto l = (__attribute__((address_space(3))) i32x4*)(unsigned int)(unsigned long long)lds_dst;
  __builtin_amdgcn_global_load_async_to_lds_b128(g, l, 0, 0);
}
#else
#define HAVE_ASYNC_LDS 0
#endif

template <int N> __device__ __forceinline__ void async_wait() {
#if HAVE_ASYNC_LDS
#if __has_builtin(__builtin_amdgcn_s_wait_asynccnt)
  __builtin_amdgcn_s_wait_asynccnt(N);
#else
  asm volatile("s_wait_asynccnt %0" :: "i"(N) : "memory");
#endif
#endif
}

// ---------------- kernel 1: xc = bf16(x - pre_encode_b) ----------------
__global__ __launch_bounds__(256) void sae_conv_x(const float* __restrict__ x,
                                                  const float* __restrict__ preb,
                                                  __bf16* __restrict__ xcb) {
  size_t i4 = (size_t)blockIdx.x * 256 + threadIdx.x;   // one float4 per thread
  f32x4 xv = ((const f32x4*)x)[i4];
  f32x4 bv = ((const f32x4*)preb)[i4 & (IN / 4 - 1)];
  f32x4 d  = xv - bv;
  bf16x4s r;
  r.v[0] = (__bf16)d.x; r.v[1] = (__bf16)d.y;
  r.v[2] = (__bf16)d.z; r.v[3] = (__bf16)d.w;
  ((bf16x4s*)xcb)[i4] = r;
}

// ---------------- kernel 2: wtt[h][k] = bf16(WT[k][h])  (LDS-tiled) ------
__global__ __launch_bounds__(256) void sae_conv_wt(const float* __restrict__ WT,
                                                   __bf16* __restrict__ wtt) {
  __shared__ float tile[32][33];
  const int h0 = blockIdx.x * 32;
  const int k0 = blockIdx.y * 32;
  const int tid = threadIdx.x;
#pragma unroll
  for (int p = 0; p < 4; ++p) {
    int lin = tid + p * 256;
    int kk = lin >> 5, hh = lin & 31;
    tile[kk][hh] = WT[(size_t)(k0 + kk) * HID + h0 + hh];   // coalesced in h
  }
  __syncthreads();
#pragma unroll
  for (int p = 0; p < 4; ++p) {
    int lin = tid + p * 256;
    int hh = lin >> 5, kk = lin & 31;
    wtt[(size_t)(h0 + hh) * IN + k0 + kk] = (__bf16)tile[kk][hh]; // coalesced in k
  }
}

// ---------------- encode helpers ----------------
// Stage one K-chunk (A: MT x KC, B: NTB x KC) into LDS.
// Exactly 5 async instructions per wave (1 A + 4 B) -> async_wait<5> drains the
// PREVIOUS chunk while these 5 stay in flight (per-wave, in-order: ISA 7.1).
__device__ __forceinline__ void stage_chunk(const __bf16* __restrict__ xcb,
                                            const __bf16* __restrict__ wtt,
                                            int t0, int ht, int kc,
                                            __bf16* As, __bf16* Bs, int tid) {
  {
    int cid = tid >> 1;                     // pair-duplicated: still 1 instr/wave
    int m = cid >> 2, ko = (cid & 3) << 3;  // 32 rows x 4 chunks of 8 bf16
    const __bf16* g = &xcb[(size_t)(t0 + m) * IN + kc + ko];
    __bf16* l = &As[m * ASTR + ko];
#if HAVE_ASYNC_LDS
    async_cp16(g, l);
#else
    *(u32x4*)l = *(const u32x4*)g;
#endif
  }
#pragma unroll
  for (int it = 0; it < 4; ++it) {
    int ch = tid + it * 256;
    int n = ch >> 2, ko = (ch & 3) << 3;    // 256 rows x 4 chunks of 8 bf16
    const __bf16* g = &wtt[(size_t)(ht + n) * IN + kc + ko];
    __bf16* l = &Bs[n * BSTR + ko];
#if HAVE_ASYNC_LDS
    async_cp16(g, l);
#else
    *(u32x4*)l = *(const u32x4*)g;
#endif
  }
}

__device__ __forceinline__ void compute_chunk(const __bf16* As, const __bf16* Bs,
                                              v8f acc[TN], int mbase, int mrow,
                                              int half8, int bkb, int ncol, int nw) {
  Frag a;                                   // lanes 0-15: K0..7/16..23 ; 16-31: K8..15/24..31
  a.q[0] = *(const u32x4*)&As[(mbase + mrow) * ASTR + half8];
  a.q[1] = *(const u32x4*)&As[(mbase + mrow) * ASTR + half8 + 16];
#pragma unroll
  for (int j = 0; j < TN; ++j) {            // one A-frag feeds 4 WMMAs
    Frag b;                                 // lanes 0-15: K0..7/16..23 ; 16-31: K8..15/24..31
    const int n = (nw * TN + j) * 16 + ncol;
    b.q[0] = *(const u32x4*)&Bs[n * BSTR + half8];
    b.q[1] = *(const u32x4*)&Bs[n * BSTR + half8 + 16];
    acc[j] = __builtin_amdgcn_wmma_f32_16x16x32_bf16(false, a.v, false, b.v,
                                                     (short)0, acc[j], false, false);
  }
}

// ---------------- kernel 3: fused encode GEMM + top-k --------------------
__global__ __launch_bounds__(256) void sae_encode_topk(const __bf16* __restrict__ xcb,
                                                       const __bf16* __restrict__ wtt,
                                                       const float*  __restrict__ b1,
                                                       float* __restrict__ ovals,
                                                       int*   __restrict__ oidx) {
  __shared__ __bf16 As2[2][MT * ASTR];                   //  5120 B (double-buffered A)
  __shared__ __align__(16) char BsLs[2 * NTB * BSTR * 2];// 40960 B (2x B / logits alias)
  __shared__ float  lval[MT][KTOP];                      //  4096 B
  __shared__ int    lidx[MT][KTOP];                      //  4096 B
  __bf16* Bs0 = (__bf16*)BsLs;
  __bf16* Bs1 = Bs0 + NTB * BSTR;
  float*  Ls  = (float*)BsLs;     // MT x LSTR logits (live only when pipe drained)

  const int tid  = threadIdx.x;
  const int lane = tid & 31;
  const int wave = tid >> 5;
  const int t0   = blockIdx.x * MT;

  // init per-token top-k lists (sorted descending)
  for (int i = tid; i < MT * KTOP; i += 256) {
    lval[i >> 5][i & 31] = -3.0e38f;
    lidx[i >> 5][i & 31] = 0;
  }

  // per-lane fragment geometry (ISA 7.12.2, wave32)
  const int mrow  = lane & 15;                 // A: M index
  const int half8 = (lane < 16) ? 0 : 8;       // A K-phase / C row offset
  const int bkb   = (lane < 16) ? 0 : 16;      // unused; B's K-phase is half8
  const int ncol  = lane & 15;                 // B/C: N index
  const int mw    = wave & 1;                  // wave's M tile (0/1)
  const int nw    = wave >> 1;                 // wave's N group (0..3)
  const int mbase = mw * 16;

  for (int ht = 0; ht < HID; ht += NTB) {
    v8f acc[TN];
#pragma unroll
    for (int j = 0; j < TN; ++j) acc[j] = (v8f){};

    // prologue: stage chunk 0
    stage_chunk(xcb, wtt, t0, ht, 0, As2[0], Bs0, tid);

    for (int kci = 0; kci < NCH; ++kci) {
      const int cur = kci & 1;
      const __bf16* Acur = As2[cur];
      const __bf16* Bcur = cur ? Bs1 : Bs0;
      if (kci + 1 < NCH) {
        // prefetch next chunk into the other buffer, keep 5 copies in flight
        stage_chunk(xcb, wtt, t0, ht, (kci + 1) * KC,
                    As2[cur ^ 1], cur ? Bs0 : Bs1, tid);
        async_wait<5>();     // drains chunk kci only; kci+1 overlaps the WMMAs below
      } else {
        async_wait<0>();     // drain pipe before Ls aliases the B buffers
      }
      __syncthreads();       // chunk kci visible to all waves
      compute_chunk(Acur, Bcur, acc, mbase, mrow, half8, bkb, ncol, nw);
      __syncthreads();       // all reads of this buffer done before it is restaged
    }

    // + b1, spill logits tile to LDS (aliased over Bs; pipe is drained)
#pragma unroll
    for (int j = 0; j < TN; ++j) {
      const int n  = (nw * TN + j) * 16 + ncol;
      const float bv = b1[ht + n];
#pragma unroll
      for (int r = 0; r < 8; ++r)
        Ls[(mbase + r + half8) * LSTR + n] = acc[j][r] + bv;
    }
    __syncthreads();
    // per-token sorted-insert top-k maintenance
    if (tid < MT) {
      for (int n = 0; n < NTB; ++n) {
        float v = Ls[tid * LSTR + n];
        if (v > lval[tid][KTOP - 1]) {
          int p = KTOP - 1;
          while (p > 0 && lval[tid][p - 1] < v) {
            lval[tid][p] = lval[tid][p - 1];
            lidx[tid][p] = lidx[tid][p - 1];
            --p;
          }
          lval[tid][p] = v;
          lidx[tid][p] = ht + n;
        }
      }
    }
    __syncthreads();
  }

  for (int i = tid; i < MT * KTOP; i += 256) {
    int m = i >> 5, j = i & 31;
    ovals[(size_t)(t0 + m) * KTOP + j] = lval[m][j];
    oidx [(size_t)(t0 + m) * KTOP + j] = lidx[m][j];
  }
}

// ---------------- kernel 4: decode  out = Σ vals*W[idx] + pre_b ----------
__global__ __launch_bounds__(256) void sae_decode(const float* __restrict__ W,
                                                  const float* __restrict__ vals,
                                                  const int*   __restrict__ tix,
                                                  const float* __restrict__ preb,
                                                  float* __restrict__ out) {
  __shared__ float sv[KTOP];
  __shared__ int   si[KTOP];
  const int t   = blockIdx.x;
  const int tid = threadIdx.x;
  if (tid < KTOP) {
    sv[tid] = vals[(size_t)t * KTOP + tid];
    si[tid] = tix [(size_t)t * KTOP + tid];
  }
  __syncthreads();
  f32x4 acc = ((const f32x4*)preb)[tid];          // 256 threads x float4 = 1024 cols
#pragma unroll 4
  for (int j = 0; j < KTOP; ++j) {
    const f32x4 w = *(const f32x4*)&W[(size_t)si[j] * IN + 4 * tid];
    acc = acc + w * sv[j];
  }
  *(f32x4*)&out[(size_t)t * IN + 4 * tid] = acc;
}

// ---------------- host launcher ----------------
extern "C" void kernel_launch(void* const* d_in, const int* in_sizes, int n_in,
                              void* d_out, int out_size, void* d_ws, size_t ws_size,
                              hipStream_t stream) {
  (void)in_sizes; (void)n_in; (void)out_size; (void)ws_size;
  const float* x    = (const float*)d_in[0];
  const float* W    = (const float*)d_in[1];
  const float* WT   = (const float*)d_in[2];
  const float* preb = (const float*)d_in[3];
  const float* b1   = (const float*)d_in[4];

  char* ws = (char*)d_ws;
  const size_t XCB_BYTES = (size_t)TOK * IN * 2;   // 32 MB bf16 xc
  const size_t WTT_BYTES = (size_t)IN * HID * 2;   // 32 MB bf16 WT^T
  const size_t VAL_BYTES = (size_t)TOK * KTOP * 4; //  2 MB
  __bf16* xcb  = (__bf16*)(ws);
  __bf16* wtt  = (__bf16*)(ws + XCB_BYTES);
  float*  vals = (float*) (ws + XCB_BYTES + WTT_BYTES);
  int*    tix  = (int*)   (ws + XCB_BYTES + WTT_BYTES + VAL_BYTES);

  sae_conv_x    <<<(TOK * IN) / 1024, 256, 0, stream>>>(x, preb, xcb);
  sae_conv_wt   <<<dim3(HID / 32, IN / 32), 256, 0, stream>>>(WT, wtt);
  sae_encode_topk<<<TOK / MT, 256, 0, stream>>>(xcb, wtt, b1, vals, tix);
  sae_decode    <<<TOK, 256, 0, stream>>>(W, vals, tix, preb, (float*)d_out);
}
